// Head_9964324127225
// MI455X (gfx1250) — matrix operand order repaired and checked by reading, not compiled, not run on hardware
//
#include <hip/hip_runtime.h>
#include <hip/hip_bf16.h>

#define B_  8
#define T_  2048
#define D_  1024
#define H_  64
#define NEGF (-1e9f)

typedef __attribute__((ext_vector_type(16))) __bf16 v16bf;
typedef __attribute__((ext_vector_type(8)))  float  v8f;

union BF16Frag {
    v16bf  v;
    uint4  q[2];
    __bf16 h[16];
};

__device__ __forceinline__ unsigned short bfbits(__bf16 b) {
    union { __bf16 b; unsigned short u; } c; c.b = b; return c.u;
}

__device__ __forceinline__ void lds_fence() {
    asm volatile("s_wait_dscnt 0" ::: "memory");
}

// ---------------------------------------------------------------------------
// Kernel 1: W (D x H, f32) -> WT (H x D, bf16) for contiguous B-fragment loads
// ---------------------------------------------------------------------------
__global__ __launch_bounds__(256) void wt_kernel(
    const float* __restrict__ Wq, const float* __restrict__ Wk,
    const float* __restrict__ Wv,
    __bf16* __restrict__ WqT, __bf16* __restrict__ WkT,
    __bf16* __restrict__ WvT) {
    int i = blockIdx.x * blockDim.x + threadIdx.x;
    if (i >= D_ * H_) return;
    int h = i & (H_ - 1);
    int d = i >> 6;
    int dst = h * D_ + d, src = d * H_ + h;
    WqT[dst] = (__bf16)Wq[src];
    WkT[dst] = (__bf16)Wk[src];
    WvT[dst] = (__bf16)Wv[src];
}

// ---------------------------------------------------------------------------
// Kernel 2: projections q,k,v = x @ W via bf16 WMMA.
// Block = 16 query rows, 4 waves = 4 N-subtiles of H=64.
// Outputs: q,k bf16 [b][t][h];  v transposed bf16 [b][h][t].
// ---------------------------------------------------------------------------
__global__ __launch_bounds__(128) void proj_kernel(
    const float* __restrict__ x,
    const __bf16* __restrict__ WqT, const __bf16* __restrict__ WkT,
    const __bf16* __restrict__ WvT,
    __bf16* __restrict__ qout, __bf16* __restrict__ kout,
    __bf16* __restrict__ vTout) {
    const int lane = threadIdx.x & 31;
    const int nt   = threadIdx.x >> 5;        // N-subtile (h block of 16)
    const int tile = blockIdx.x;              // 0 .. B*T/16-1
    const int b    = tile >> 7;               // T/16 = 128 tiles per batch
    const int rowbase = (tile & 127) << 4;
    const int g  = lane >> 4;
    const int nn = lane & 15;

    const float* xrow = x + ((size_t)b * T_ + rowbase + nn) * (size_t)D_;
    const int hN = nt * 16 + nn;
    const __bf16* wq = WqT + (size_t)hN * D_;
    const __bf16* wk = WkT + (size_t)hN * D_;
    const __bf16* wv = WvT + (size_t)hN * D_;

    v8f accq = {}, acck = {}, accv = {};

    for (int k0 = 0; k0 < D_; k0 += 32) {
        // A fragment: 16-bit A layout, halves 0..7 -> K=k0+8g.., 8..15 -> K=k0+16+8g..
        BF16Frag a;
        const float* p1 = xrow + k0 + 8 * g;
        const float* p2 = xrow + k0 + 16 + 8 * g;
#pragma unroll
        for (int e = 0; e < 8; ++e) a.h[e]     = (__bf16)p1[e];
#pragma unroll
        for (int e = 0; e < 8; ++e) a.h[8 + e] = (__bf16)p2[e];

        // B fragments: lane N=nn, halves 0..7 -> K=k0+8g.., 8..15 -> K=k0+16+8g..
        BF16Frag bq, bk, bv;
        const uint4* s;
        s = (const uint4*)(wq + k0 + 8 * g); bq.q[0] = s[0]; bq.q[1] = s[2];
        s = (const uint4*)(wk + k0 + 8 * g); bk.q[0] = s[0]; bk.q[1] = s[2];
        s = (const uint4*)(wv + k0 + 8 * g); bv.q[0] = s[0]; bv.q[1] = s[2];

        accq = __builtin_amdgcn_wmma_f32_16x16x32_bf16(false, a.v, false, bq.v,
                                                       (short)0, accq, false, false);
        acck = __builtin_amdgcn_wmma_f32_16x16x32_bf16(false, a.v, false, bk.v,
                                                       (short)0, acck, false, false);
        accv = __builtin_amdgcn_wmma_f32_16x16x32_bf16(false, a.v, false, bv.v,
                                                       (short)0, accv, false, false);
    }

    // C layout: M = r + 8g, N = nn
#pragma unroll
    for (int r = 0; r < 8; ++r) {
        int tr = rowbase + r + 8 * g;
        size_t qi = ((size_t)b * T_ + tr) * H_ + hN;
        qout[qi] = (__bf16)accq[r];
        kout[qi] = (__bf16)acck[r];
        vTout[((size_t)b * H_ + hN) * T_ + tr] = (__bf16)accv[r];
    }
}

// ---------------------------------------------------------------------------
// Kernel 3: flash attention. One wave32 per 16-query tile; 32-key tiles.
// Wave-private LDS slices; no workgroup barriers (divergent trip counts).
// WPB=2 -> 512 small blocks so the scheduler can balance the causal
// (qbase-proportional) per-tile work across WGPs.
// ---------------------------------------------------------------------------
#define WPB 2
__global__ __launch_bounds__(WPB * 32) void attn_kernel(
    const __bf16* __restrict__ qm, const __bf16* __restrict__ km,
    const __bf16* __restrict__ vT, const int* __restrict__ vlens,
    float* __restrict__ out) {
    __shared__ __attribute__((aligned(16))) float   Sbuf[WPB][16][32];
    __shared__ __attribute__((aligned(16))) __bf16  Pbuf[WPB][16][32];
    __shared__ __attribute__((aligned(16))) float   Astat[WPB][16];
    __shared__ __attribute__((aligned(16))) float   Lstat[WPB][16];

    const int lane = threadIdx.x & 31;
    const int wave = threadIdx.x >> 5;
    const int tile = blockIdx.x * WPB + wave;     // 0 .. B*T/16-1
    const int b     = tile >> 7;
    const int qbase = (tile & 127) << 4;
    const int g  = lane >> 4;
    const int nn = lane & 15;
    const int vlen = vlens[b];
    const float scale = 0.125f;                   // 1/sqrt(64)

    // Q fragments (A layout), K-chunks h=0..31 and 32..63
    BF16Frag qa[2];
    {
        const __bf16* qrow = qm + ((size_t)b * T_ + qbase + nn) * H_;
#pragma unroll
        for (int f = 0; f < 2; ++f) {
            qa[f].q[0] = *(const uint4*)(qrow + 32 * f + 8 * g);
            qa[f].q[1] = *(const uint4*)(qrow + 32 * f + 16 + 8 * g);
        }
    }

    v8f o[4] = {v8f{}, v8f{}, v8f{}, v8f{}};
    float mrow = -1e30f, lrow = 0.0f;             // owner-lane running stats
    const int srow  = lane >> 1;                  // softmax: 2 lanes per row
    const int scol0 = (lane & 1) * 16;

    const int nEnd = qbase + 16;                  // causal key bound (exclusive)

    for (int n0 = 0; n0 < nEnd && n0 < vlen; n0 += 32) {
        // ---- S = (q k^T) * scale, masked, -> LDS ----
#pragma unroll
        for (int s = 0; s < 2; ++s) {
            const __bf16* krow = km + ((size_t)b * T_ + n0 + 16 * s + nn) * H_;
            BF16Frag kb0, kb1;
            kb0.q[0] = *(const uint4*)(krow + 8 * g);
            kb0.q[1] = *(const uint4*)(krow + 8 * g + 16);
            kb1.q[0] = *(const uint4*)(krow + 32 + 8 * g);
            kb1.q[1] = *(const uint4*)(krow + 32 + 8 * g + 16);
            v8f sacc = {};
            sacc = __builtin_amdgcn_wmma_f32_16x16x32_bf16(false, qa[0].v, false, kb0.v,
                                                           (short)0, sacc, false, false);
            sacc = __builtin_amdgcn_wmma_f32_16x16x32_bf16(false, qa[1].v, false, kb1.v,
                                                           (short)0, sacc, false, false);
            const int nk = n0 + 16 * s + nn;
#pragma unroll
            for (int r = 0; r < 8; ++r) {
                int tq = qbase + r + 8 * g;
                bool ok = (nk <= tq) && (nk < vlen) && (tq < vlen);
                Sbuf[wave][r + 8 * g][16 * s + nn] = ok ? sacc[r] * scale : NEGF;
            }
        }
        lds_fence();

        // ---- online softmax: 2 lanes per row, 16 cols each ----
        float sv[16];
        {
            const float4* sp = (const float4*)&Sbuf[wave][srow][scol0];
#pragma unroll
            for (int i = 0; i < 4; ++i) {
                float4 f = sp[i];
                sv[4 * i + 0] = f.x; sv[4 * i + 1] = f.y;
                sv[4 * i + 2] = f.z; sv[4 * i + 3] = f.w;
            }
        }
        float tmax = sv[0];
#pragma unroll
        for (int i = 1; i < 16; ++i) tmax = fmaxf(tmax, sv[i]);
        tmax = fmaxf(tmax, __shfl_xor(tmax, 1, 32));
        float mnew  = fmaxf(mrow, tmax);
        float alpha = __expf(mrow - mnew);
        float tsum = 0.0f;
        unsigned int pw[8];
#pragma unroll
        for (int i = 0; i < 8; ++i) {
            float p0 = __expf(sv[2 * i]     - mnew);
            float p1 = __expf(sv[2 * i + 1] - mnew);
            tsum += p0 + p1;
            pw[i] = (unsigned int)bfbits((__bf16)p0) |
                    ((unsigned int)bfbits((__bf16)p1) << 16);
        }
        tsum += __shfl_xor(tsum, 1, 32);
        lrow = lrow * alpha + tsum;
        mrow = mnew;
        {
            uint4* pp = (uint4*)&Pbuf[wave][srow][scol0];
            pp[0] = make_uint4(pw[0], pw[1], pw[2], pw[3]);
            pp[1] = make_uint4(pw[4], pw[5], pw[6], pw[7]);
        }
        if ((lane & 1) == 0) Astat[wave][srow] = alpha;
        lds_fence();

        // ---- rescale O, then O += P @ V ----
#pragma unroll
        for (int r = 0; r < 8; ++r) {
            float al = Astat[wave][r + 8 * g];
#pragma unroll
            for (int c = 0; c < 4; ++c) o[c][r] *= al;
        }
        BF16Frag pa;   // P as A-matrix fragment
        pa.q[0] = *(const uint4*)&Pbuf[wave][nn][8 * g];
        pa.q[1] = *(const uint4*)&Pbuf[wave][nn][16 + 8 * g];
#pragma unroll
        for (int c = 0; c < 4; ++c) {
            const __bf16* vrow =
                vT + ((size_t)b * H_ + 16 * c + nn) * T_ + n0 + 8 * g;
            BF16Frag vb;
            vb.q[0] = *(const uint4*)(vrow);
            vb.q[1] = *(const uint4*)(vrow + 16);
            o[c] = __builtin_amdgcn_wmma_f32_16x16x32_bf16(false, pa.v, false, vb.v,
                                                           (short)0, o[c], false, false);
        }
    }

    // ---- epilogue: divide by l, zero fully-masked rows ----
    if ((lane & 1) == 0) Lstat[wave][srow] = lrow;
    lds_fence();
#pragma unroll
    for (int r = 0; r < 8; ++r) {
        int tq = qbase + r + 8 * g;
        float linv = 0.0f;
        if (tq < vlen) linv = 1.0f / Lstat[wave][r + 8 * g];
#pragma unroll
        for (int c = 0; c < 4; ++c) {
            out[((size_t)b * T_ + tq) * H_ + 16 * c + nn] = o[c][r] * linv;
        }
    }
}

// ---------------------------------------------------------------------------
extern "C" void kernel_launch(void* const* d_in, const int* in_sizes, int n_in,
                              void* d_out, int out_size, void* d_ws, size_t ws_size,
                              hipStream_t stream) {
    const float* x    = (const float*)d_in[0];
    const float* Wq   = (const float*)d_in[1];
    const float* Wk   = (const float*)d_in[2];
    const float* Wv   = (const float*)d_in[3];
    const int*   vls  = (const int*)d_in[4];
    float*       outp = (float*)d_out;

    char* ws = (char*)d_ws;
    const size_t wt_bytes  = (size_t)H_ * D_ * 2;       // 128 KB each
    const size_t qkv_bytes = (size_t)B_ * T_ * H_ * 2;  // 2 MB each
    __bf16* WqT = (__bf16*)ws;             ws += wt_bytes;
    __bf16* WkT = (__bf16*)ws;             ws += wt_bytes;
    __bf16* WvT = (__bf16*)ws;             ws += wt_bytes;
    __bf16* qb  = (__bf16*)ws;             ws += qkv_bytes;
    __bf16* kb  = (__bf16*)ws;             ws += qkv_bytes;
    __bf16* vTb = (__bf16*)ws;             ws += qkv_bytes;

    wt_kernel<<<(D_ * H_ + 255) / 256, 256, 0, stream>>>(Wq, Wk, Wv, WqT, WkT, WvT);
    proj_kernel<<<B_ * T_ / 16, 128, 0, stream>>>(x, WqT, WkT, WvT, qb, kb, vTb);
    attn_kernel<<<(B_ * T_ / 16) / WPB, WPB * 32, 0, stream>>>(qb, kb, vTb, vls, outp);
}
